// RBFKernel_65481071402880
// MI455X (gfx1250) — matrix operand order repaired and checked by reading, not compiled, not run on hardware
//
#include <hip/hip_runtime.h>

// ---- types for WMMA ----
typedef __attribute__((ext_vector_type(16))) __bf16 v16bf;
typedef __attribute__((ext_vector_type(8)))  __bf16 v8bf;
typedef __attribute__((ext_vector_type(8)))  float  v8f;

#define D      256      // feature dim
#define NROW   8192
#define NCOL   8192
#define BM     128      // block tile M
#define BN     128      // block tile N
#define KT     32       // K chunk per WMMA pass
#define NK     (D / KT) // 8 K-chunks
#define LDS_RS 40       // shorts per LDS row: 32 data + 8 pad (80 B => conflict-free b128 loads)
#define TILE_SH   (BM * LDS_RS)          // 5120 shorts / 10240 B per tile
#define STAGE_SH  (4 * TILE_SH)          // 20480 shorts / 40960 B per stage (xh,xl,yh,yl)
#define SMEM_BYTES (2 * STAGE_SH * 2)    // 81920 B double-buffered

// fp32 -> bf16 round-to-nearest-even in integer VALU
__device__ __forceinline__ unsigned short f2bf_rne(float f) {
    unsigned u = __float_as_uint(f);
    u += 0x7FFFu + ((u >> 16) & 1u);
    return (unsigned short)(u >> 16);
}
__device__ __forceinline__ float bf2f(unsigned short h) {
    return __uint_as_float(((unsigned)h) << 16);
}

// per-lane 16B global->LDS async DMA (gfx1250), SVS addressing: saddr base + 32b vgpr offset
__device__ __forceinline__ void async_copy16(unsigned lds_off, const unsigned short* base,
                                             unsigned goff_bytes) {
    asm volatile("global_load_async_to_lds_b128 %0, %1, %2"
                 :: "v"(lds_off), "v"(goff_bytes), "s"(base) : "memory");
}

// -------- pre-pass 1: row squared norms of x and y --------
__global__ void __launch_bounds__(256)
norms_kernel(const float* __restrict__ x, const float* __restrict__ y,
             float* __restrict__ ns) {
    int row = blockIdx.x * blockDim.x + threadIdx.x;   // 0 .. 16383
    if (row >= 2 * NROW) return;
    const float4* p = (const float4*)((row < NROW) ? (x + (size_t)row * D)
                                                   : (y + (size_t)(row - NROW) * D));
    float s = 0.f;
#pragma unroll 8
    for (int i = 0; i < D / 4; ++i) {
        float4 v = p[i];
        s += v.x * v.x + v.y * v.y + v.z * v.z + v.w * v.w;
    }
    ns[row] = s;
}

// -------- pre-pass 2: one-shot fp32 -> bf16 hi/lo split (bf16x3 operands) --------
__global__ void __launch_bounds__(256)
convert_kernel(const float* __restrict__ X, const float* __restrict__ Y,
               unsigned short* __restrict__ Xh, unsigned short* __restrict__ Xl,
               unsigned short* __restrict__ Yh, unsigned short* __restrict__ Yl) {
    const int half = NROW * D / 4;                     // float4 count per matrix
    int t = blockIdx.x * 256 + threadIdx.x;            // 0 .. 2*half-1
    const float* src;
    unsigned short *H, *L;
    int i = t;
    if (t < half)  { src = X; H = Xh; L = Xl; }
    else           { src = Y; H = Yh; L = Yl; i = t - half; }
    float4 v = ((const float4*)src)[i];
    unsigned short h0 = f2bf_rne(v.x), h1 = f2bf_rne(v.y),
                   h2 = f2bf_rne(v.z), h3 = f2bf_rne(v.w);
    unsigned short l0 = f2bf_rne(v.x - bf2f(h0)), l1 = f2bf_rne(v.y - bf2f(h1)),
                   l2 = f2bf_rne(v.z - bf2f(h2)), l3 = f2bf_rne(v.w - bf2f(h3));
    uint2 u;
    u.x = (unsigned)h0 | ((unsigned)h1 << 16);
    u.y = (unsigned)h2 | ((unsigned)h3 << 16);
    *(uint2*)(H + (size_t)i * 4) = u;
    u.x = (unsigned)l0 | ((unsigned)l1 << 16);
    u.y = (unsigned)l2 | ((unsigned)l3 << 16);
    *(uint2*)(L + (size_t)i * 4) = u;
}

// issue one stage: 128x32 bf16 of xh,xl,yh,yl -> LDS via async DMA (8 ops / thread)
__device__ __forceinline__ void issue_stage(unsigned lds0,
                                            const unsigned short* Xh, const unsigned short* Xl,
                                            const unsigned short* Yh, const unsigned short* Yl,
                                            int i0, int j0, int kt, int tid) {
#pragma unroll
    for (int q = 0; q < 2; ++q) {
        int c   = tid + q * 256;            // 0..511 : 16B segments of a 8 KB tile
        int r   = c >> 2;                   // tile row 0..127
        int seg = c & 3;                    // 16B segment within 64B row
        unsigned gx = (unsigned)((i0 + r) * (D * 2) + kt * (KT * 2) + seg * 16);
        unsigned gy = (unsigned)((j0 + r) * (D * 2) + kt * (KT * 2) + seg * 16);
        unsigned l  = lds0 + (unsigned)(r * (LDS_RS * 2) + seg * 16);
        async_copy16(l,                     Xh, gx);
        async_copy16(l + 1 * TILE_SH * 2,   Xl, gx);
        async_copy16(l + 2 * TILE_SH * 2,   Yh, gy);
        async_copy16(l + 3 * TILE_SH * 2,   Yl, gy);
    }
}

// -------- main kernel: double-buffered async staging + bf16x3 WMMA + RBF epilogue --------
__global__ void __launch_bounds__(256)
rbf_wmma_kernel(const unsigned short* __restrict__ Xh, const unsigned short* __restrict__ Xl,
                const unsigned short* __restrict__ Yh, const unsigned short* __restrict__ Yl,
                const float* __restrict__ xs, const float* __restrict__ ys,
                float* __restrict__ out) {
    extern __shared__ __align__(16) unsigned short smem[];   // 2 stages x 40960 B

    const int tid  = threadIdx.x;
    const int wave = tid >> 5;
    const int lane = tid & 31;
    const int l15  = lane & 15;
    const int lhi  = lane >> 4;
    const int i0   = blockIdx.y * BM;
    const int j0   = blockIdx.x * BN;
    const unsigned lbase = (unsigned)(uintptr_t)smem;         // LDS byte offset of buffer

    v8f acc[8] = {};

    // prologue: stage 0 <- chunk 0
    issue_stage(lbase, Xh, Xl, Yh, Yl, i0, j0, 0, tid);

#pragma unroll
    for (int kt = 0; kt < NK; ++kt) {
        if (kt + 1 < NK) {
            // overlap: DMA chunk kt+1 while chunk kt's WMMAs run
            issue_stage(lbase + (unsigned)(((kt + 1) & 1) * STAGE_SH * 2),
                        Xh, Xl, Yh, Yl, i0, j0, kt + 1, tid);
            asm volatile("s_wait_asynccnt 0x8" ::: "memory");  // chunk kt landed (in-order)
        } else {
            asm volatile("s_wait_asynccnt 0x0" ::: "memory");
        }
        __syncthreads();   // all waves' DMA for chunk kt visible

        const unsigned short* st = smem + (size_t)((kt & 1) * STAGE_SH);
        const unsigned short* txh = st;
        const unsigned short* txl = st + 1 * TILE_SH;
        const unsigned short* tyh = st + 2 * TILE_SH;
        const unsigned short* tyl = st + 3 * TILE_SH;

        // A fragments (16x32 bf16): lanes 0-15 row M=l hold K 0-7,16-23; lanes 16-31 K 8-15,24-31
        const int am = wave * 16 + l15;
        const unsigned short* axh = txh + am * LDS_RS;
        const unsigned short* axl = txl + am * LDS_RS;
        v8bf ah0 = *(const v8bf*)(axh + lhi * 8);
        v8bf ah1 = *(const v8bf*)(axh + 16 + lhi * 8);
        v8bf al0 = *(const v8bf*)(axl + lhi * 8);
        v8bf al1 = *(const v8bf*)(axl + 16 + lhi * 8);
        v16bf a_hi = __builtin_shufflevector(ah0, ah1, 0,1,2,3,4,5,6,7,8,9,10,11,12,13,14,15);
        v16bf a_lo = __builtin_shufflevector(al0, al1, 0,1,2,3,4,5,6,7,8,9,10,11,12,13,14,15);

#pragma unroll
        for (int nt = 0; nt < 8; ++nt) {
            // B fragment (32x16 bf16): lanes 0-15 col N=l hold K 0-7,16-23; lanes 16-31 K 8-15,24-31
            const int bn = nt * 16 + l15;
            const unsigned short* byh = tyh + bn * LDS_RS;
            const unsigned short* byl = tyl + bn * LDS_RS;
            v8bf bh0 = *(const v8bf*)(byh + lhi * 8);
            v8bf bh1 = *(const v8bf*)(byh + 16 + lhi * 8);
            v8bf bl0 = *(const v8bf*)(byl + lhi * 8);
            v8bf bl1 = *(const v8bf*)(byl + 16 + lhi * 8);
            v16bf b_hi = __builtin_shufflevector(bh0, bh1, 0,1,2,3,4,5,6,7,8,9,10,11,12,13,14,15);
            v16bf b_lo = __builtin_shufflevector(bl0, bl1, 0,1,2,3,4,5,6,7,8,9,10,11,12,13,14,15);

            // bf16x3: x.y ~= xh*yh + xh*yl + xl*yh
            acc[nt] = __builtin_amdgcn_wmma_f32_16x16x32_bf16(
                false, a_hi, false, b_hi, (short)0, acc[nt], false, false);
            acc[nt] = __builtin_amdgcn_wmma_f32_16x16x32_bf16(
                false, a_hi, false, b_lo, (short)0, acc[nt], false, false);
            acc[nt] = __builtin_amdgcn_wmma_f32_16x16x32_bf16(
                false, a_lo, false, b_hi, (short)0, acc[nt], false, false);
        }
        __syncthreads();   // stage reusable: next iteration's DMA may overwrite it
    }

    // ---- epilogue: d2 = xsq + ysq - 2*dot, out = exp(-max(d2,0))  (GAMMA = 1) ----
    float xsq[8];
#pragma unroll
    for (int v = 0; v < 8; ++v)
        xsq[v] = xs[i0 + wave * 16 + (lhi << 3) + v];

#pragma unroll
    for (int nt = 0; nt < 8; ++nt) {
        const int j = j0 + nt * 16 + l15;
        const float ysq = ys[j];
#pragma unroll
        for (int v = 0; v < 8; ++v) {
            const int m = i0 + wave * 16 + (lhi << 3) + v;
            float d2 = xsq[v] + ysq - 2.0f * acc[nt][v];
            d2 = fmaxf(d2, 0.0f);
            out[(size_t)m * NCOL + j] = __expf(-d2);
        }
    }
}

extern "C" void kernel_launch(void* const* d_in, const int* in_sizes, int n_in,
                              void* d_out, int out_size, void* d_ws, size_t ws_size,
                              hipStream_t stream) {
    const float* x = (const float*)d_in[0];
    const float* y = (const float*)d_in[1];
    float* out = (float*)d_out;

    // workspace layout: Xh|Xl|Yh|Yl (bf16, 4 MiB each) then xs|ys (fp32 norms)
    unsigned short* Xh = (unsigned short*)d_ws;
    unsigned short* Xl = Xh + (size_t)NROW * D;
    unsigned short* Yh = Xl + (size_t)NROW * D;
    unsigned short* Yl = Yh + (size_t)NROW * D;
    float* xs = (float*)(Yl + (size_t)NROW * D);
    float* ys = xs + NROW;

    norms_kernel<<<(2 * NROW + 255) / 256, 256, 0, stream>>>(x, y, xs);
    convert_kernel<<<(2 * NROW * D / 4 + 255) / 256, 256, 0, stream>>>(x, y, Xh, Xl, Yh, Yl);

    dim3 grid(NCOL / BN, NROW / BM);
    rbf_wmma_kernel<<<grid, 256, SMEM_BYTES, stream>>>(Xh, Xl, Yh, Yl, xs, ys, out);
}
